// BitAttention_16716012716235
// MI455X (gfx1250) — 16-bit and 8-bit matrix operand order repaired and checked by reading, compiled, not run on hardware
//
#include <hip/hip_runtime.h>
#include <hip/hip_fp16.h>
#include <stdint.h>

#define B_     2
#define T_     2048
#define D_     2048
#define H_     16
#define HD_    128
#define NROWS  (B_ * T_)
#define EPS_   1e-5f
#define TOPK_K 1126

typedef __attribute__((ext_vector_type(8)))  int       v8i;
typedef __attribute__((ext_vector_type(16))) int       v16i;
typedef __attribute__((ext_vector_type(8)))  float     v8f;
typedef __attribute__((ext_vector_type(16))) _Float16  v16h;

__constant__ uint8_t FP8_TAB[17] = {
  0xD0, 0xCE, 0xCC, 0xCA, 0xC8, 0xC4, 0xC0, 0xB8,
  0x00,
  0x38, 0x40, 0x44, 0x48, 0x4A, 0x4C, 0x4E, 0x50
};

__device__ __forceinline__ v8f wmma_fp8(v16i a, v16i b, v8f c) {
  return __builtin_amdgcn_wmma_f32_16x16x128_fp8_fp8(a, b, (short)0, c, false, false);
}
__device__ __forceinline__ v8i wmma_iu8(v8i a, v8i b, v8i c) {
  return __builtin_amdgcn_wmma_i32_16x16x64_iu8(true, a, true, b, c, false, false);
}
__device__ __forceinline__ v8f wmma_f16(v16h a, v16h b, v8f c) {
  return __builtin_amdgcn_wmma_f32_16x16x32_f16(false, a, false, b, (short)0, c,
                                                false, false);
}

__device__ __forceinline__ v8i fragA8(const int8_t* base, int ld, int row0, int k0) {
  int lane = threadIdx.x & 31;
  int m  = row0 + (lane & 15);
  int hf = lane >> 4;
  const int8_t* r = base + (size_t)m * ld + k0 + hf * 16;
  union { long long q[4]; v8i v; } u;
  u.q[0] = *(const long long*)(r +  0);
  u.q[1] = *(const long long*)(r +  8);
  u.q[2] = *(const long long*)(r + 32);
  u.q[3] = *(const long long*)(r + 40);
  return u.v;
}
__device__ __forceinline__ v16i fragA8_128(const uint8_t* base, int ld, int row0, int k0) {
  int lane = threadIdx.x & 31;
  int m  = row0 + (lane & 15);
  int hf = lane >> 4;
  const uint8_t* r = base + (size_t)m * ld + k0 + hf * 16;
  union { long long q[8]; v16i v; } u;
  u.q[0] = *(const long long*)(r +   0);
  u.q[1] = *(const long long*)(r +   8);
  u.q[2] = *(const long long*)(r +  32);
  u.q[3] = *(const long long*)(r +  40);
  u.q[4] = *(const long long*)(r +  64);
  u.q[5] = *(const long long*)(r +  72);
  u.q[6] = *(const long long*)(r +  96);
  u.q[7] = *(const long long*)(r + 104);
  return u.v;
}
__device__ __forceinline__ v8i fragB8(const int8_t* base, int ld, int col0, int k0) {
  int lane = threadIdx.x & 31;
  int n  = col0 + (lane & 15);
  int hf = lane >> 4;
  const int8_t* r = base + (size_t)n * ld + k0 + hf * 16;
  union { float4 f[2]; v8i v; } u;
  u.f[0] = *(const float4*)(r +  0);
  u.f[1] = *(const float4*)(r + 32);
  return u.v;
}
__device__ __forceinline__ v16i fragB8_128(const uint8_t* base, int ld, int col0, int k0) {
  int lane = threadIdx.x & 31;
  int n  = col0 + (lane & 15);
  int hf = lane >> 4;
  const uint8_t* r = base + (size_t)n * ld + k0 + hf * 16;
  union { float4 f[4]; v16i v; } u;
  u.f[0] = *(const float4*)(r +  0);
  u.f[1] = *(const float4*)(r + 32);
  u.f[2] = *(const float4*)(r + 64);
  u.f[3] = *(const float4*)(r + 96);
  return u.v;
}
__device__ __forceinline__ v16h fragA16(const __half* base, int ld, int row0, int k0) {
  int lane = threadIdx.x & 31;
  int m  = row0 + (lane & 15);
  int hf = lane >> 4;
  const __half* r = base + (size_t)m * ld + k0 + hf * 8;
  union { float4 f[2]; v16h v; } u;
  u.f[0] = *(const float4*)(r);
  u.f[1] = *(const float4*)(r + 16);
  return u.v;
}
__device__ __forceinline__ v16h fragB16(const __half* base, int ld, int col0, int k0) {
  int lane = threadIdx.x & 31;
  int n  = col0 + (lane & 15);
  int hf = lane >> 4;
  const __half* r = base + (size_t)n * ld + k0 + hf * 8;
  union { float4 f[2]; v16h v; } u;
  u.f[0] = *(const float4*)(r);
  u.f[1] = *(const float4*)(r + 16);
  return u.v;
}

__global__ void k_zero(float* p, int n) {
  int i = blockIdx.x * blockDim.x + threadIdx.x;
  if (i < n) p[i] = 0.f;
}

__global__ __launch_bounds__(256)
void k_abssum(const float* w0, const float* w1, const float* w2, const float* w3,
              float* sums) {
  const float* w = (blockIdx.z == 0) ? w0 : (blockIdx.z == 1) ? w1
                 : (blockIdx.z == 2) ? w2 : w3;
  __shared__ float red[256];
  int t = threadIdx.x;
  float s = 0.f;
  for (size_t i = (size_t)blockIdx.x * blockDim.x + t; i < (size_t)D_ * D_;
       i += (size_t)gridDim.x * blockDim.x)
    s += fabsf(w[i]);
  red[t] = s; __syncthreads();
  for (int st = 128; st > 0; st >>= 1) {
    if (t < st) red[t] += red[t + st];
    __syncthreads();
  }
  if (t == 0) atomicAdd(&sums[blockIdx.z], red[0]);
}

__global__ __launch_bounds__(256)
void k_ternarize(const float* w0, const float* w1, const float* w2, const float* w3,
                 const float* sums, uint8_t* wt_fp8, int8_t* wt_i8, float* wscale) {
  int z = blockIdx.z;
  const float* w = (z == 0) ? w0 : (z == 1) ? w1 : (z == 2) ? w2 : w3;
  uint8_t* dst8 = wt_fp8 + (size_t)z * D_ * D_;
  int8_t*  dsti = wt_i8  + (size_t)z * D_ * D_;
  float scale = sums[z] / (float)((size_t)D_ * D_);
  float inv = 1.f / (scale + EPS_);
  for (size_t i = (size_t)blockIdx.x * blockDim.x + threadIdx.x; i < (size_t)D_ * D_;
       i += (size_t)gridDim.x * blockDim.x) {
    int q = (int)fminf(1.f, fmaxf(-1.f, roundf(w[i] * inv)));
    dst8[i] = FP8_TAB[q + 8];
    dsti[i] = (int8_t)q;
  }
  if (blockIdx.x == 0 && threadIdx.x == 0) wscale[z] = scale;
}

__global__ __launch_bounds__(256)
void k_quant_x(const float* x, uint8_t* xq, float* rscale) {
  int row = blockIdx.x;
  int t = threadIdx.x;
  const float* xr = x + (size_t)row * D_;
  __shared__ float red[256];
  float mx = 0.f;
  for (int i = t; i < D_; i += 256) mx = fmaxf(mx, fabsf(xr[i]));
  red[t] = mx; __syncthreads();
  for (int st = 128; st > 0; st >>= 1) {
    if (t < st) red[t] = fmaxf(red[t], red[t + st]);
    __syncthreads();
  }
  float amax = fmaxf(red[0], EPS_);
  float s = 7.f / amax;
  for (int i = t; i < D_; i += 256) {
    int q = (int)fminf(7.f, fmaxf(-8.f, roundf(xr[i] * s)));
    xq[(size_t)row * D_ + i] = FP8_TAB[q + 8];
  }
  if (t == 0) rscale[row] = 1.f / s;
}

__global__ __launch_bounds__(256)
void k_gemm_qkv(const uint8_t* xq, const float* rscale, const uint8_t* wt,
                const float* wscale, __half* qh, __half* kh, __half* vh) {
  int z = blockIdx.z;
  const uint8_t* w = wt + (size_t)z * D_ * D_;
  __half* out = (z == 0) ? qh : (z == 1) ? kh : vh;
  float ws = wscale[z];
  int wave = threadIdx.x >> 5;
  int lane = threadIdx.x & 31;
  int m0 = blockIdx.y * 16;
  int n0 = blockIdx.x * 128 + wave * 16;
  v8f acc = {};
  for (int k0 = 0; k0 < D_; k0 += 128) {
    v16i a = fragA8_128(xq, D_, m0, k0);
    v16i b = fragB8_128(w,  D_, n0, k0);
    acc = wmma_fp8(a, b, acc);
  }
  int hf = lane >> 4;
  int n = n0 + (lane & 15);
  for (int r = 0; r < 8; ++r) {
    int m = m0 + r + 8 * hf;
    float v = acc[r] * rscale[m] * ws;
    out[(size_t)m * D_ + n] = __float2half(v);
  }
}

__global__ __launch_bounds__(256)
void k_rope(__half* qh, __half* kh) {
  size_t i = (size_t)blockIdx.x * blockDim.x + threadIdx.x;
  __half* buf = (blockIdx.y == 0) ? qh : kh;
  int d = (int)(i & 63);
  int h = (int)((i >> 6) & 15);
  int n = (int)(i >> 10);
  int t = n % T_;
  float inv = __powf(10000.f, -((float)(2 * d)) / (float)HD_);
  float fr = (float)t * inv;
  float c = __cosf(fr), s = __sinf(fr);
  size_t base = (size_t)n * D_ + h * HD_;
  float x1 = __half2float(buf[base + d]);
  float x2 = __half2float(buf[base + d + 64]);
  buf[base + d]      = __float2half(x1 * c - x2 * s);
  buf[base + d + 64] = __float2half(x2 * c + x1 * s);
}

__global__ __launch_bounds__(128)
void k_attn(const __half* qh, const __half* kh, const __half* vh, __half* oh) {
  int bh = blockIdx.x;
  int b = bh / H_, h = bh % H_;
  int wave = threadIdx.x >> 5;
  int lane = threadIdx.x & 31;
  int hf = lane >> 4;
  int qt = blockIdx.y * 64 + wave * 16;

  const __half* Q = qh + (size_t)b * T_ * D_ + h * HD_;
  const __half* K = kh + (size_t)b * T_ * D_ + h * HD_;
  const __half* V = vh + (size_t)b * T_ * D_ + h * HD_;

  __shared__ __half Vt[HD_][40];
  __shared__ __half Pl[4][16][32];

  v16h qa[4];
  for (int c = 0; c < 4; ++c) qa[c] = fragA16(Q, D_, qt, c * 32);

  v8f o[8] = {};
  float rmax[8], rsum[8];
  for (int r = 0; r < 8; ++r) { rmax[r] = -1e30f; rsum[r] = 0.f; }
  const float sm_scale = 0.08838834764831845f;

  for (int kt = 0; kt < T_; kt += 32) {
    __syncthreads();
    for (int idx = threadIdx.x; idx < 32 * HD_; idx += 128) {
      int key = idx >> 7, d = idx & 127;
      Vt[d][key] = V[(size_t)(kt + key) * D_ + d];
    }
    __syncthreads();

    v8f s0 = {}, s1 = {};
    for (int c = 0; c < 4; ++c) {
      v16h kb0 = fragB16(K, D_, kt +  0, c * 32);
      v16h kb1 = fragB16(K, D_, kt + 16, c * 32);
      s0 = wmma_f16(qa[c], kb0, s0);
      s1 = wmma_f16(qa[c], kb1, s1);
    }

    for (int r = 0; r < 8; ++r) {
      float a0 = s0[r] * sm_scale;
      float a1 = s1[r] * sm_scale;
      float v = fmaxf(a0, a1);
      for (int off = 8; off >= 1; off >>= 1) v = fmaxf(v, __shfl_xor(v, off, 32));
      float nm = fmaxf(rmax[r], v);
      float alpha = __expf(rmax[r] - nm);
      rmax[r] = nm;
      float p0 = __expf(a0 - nm);
      float p1 = __expf(a1 - nm);
      float ps = p0 + p1;
      for (int off = 8; off >= 1; off >>= 1) ps += __shfl_xor(ps, off, 32);
      rsum[r] = rsum[r] * alpha + ps;
      for (int c = 0; c < 8; ++c) o[c][r] = o[c][r] * alpha;
      Pl[wave][r + 8 * hf][lane & 15]        = __float2half(p0);
      Pl[wave][r + 8 * hf][16 + (lane & 15)] = __float2half(p1);
    }
    __syncthreads();

    v16h pa = fragA16((const __half*)&Pl[wave][0][0], 32, 0, 0);
    for (int c = 0; c < 8; ++c) {
      v16h vb = fragB16((const __half*)&Vt[0][0], 40, c * 16, 0);
      o[c] = wmma_f16(pa, vb, o[c]);
    }
  }

  for (int r = 0; r < 8; ++r) {
    float inv = 1.f / rsum[r];
    int tq = qt + r + 8 * hf;
    size_t rowbase = ((size_t)b * T_ + tq) * D_ + h * HD_;
    for (int c = 0; c < 8; ++c)
      oh[rowbase + c * 16 + (lane & 15)] = __float2half(o[c][r] * inv);
  }
}

__global__ __launch_bounds__(256)
void k_topk_quant(const __half* oh, int8_t* oq, float* rscale_o) {
  int row = blockIdx.x;
  int t = threadIdx.x;
  const __half* xr = oh + (size_t)row * D_;
  __shared__ float cache[D_];
  __shared__ float redf[256];
  __shared__ int   redi[256];
  float mx = 0.f;
  for (int i = t; i < D_; i += 256) {
    float a = fabsf(__half2float(xr[i]));
    cache[i] = a;
    mx = fmaxf(mx, a);
  }
  redf[t] = mx; __syncthreads();
  for (int st = 128; st > 0; st >>= 1) {
    if (t < st) redf[t] = fmaxf(redf[t], redf[t + st]);
    __syncthreads();
  }
  float rowmax = redf[0];
  float lo = 0.f, hi = rowmax;
  for (int it = 0; it < 20; ++it) {
    float th = 0.5f * (lo + hi);
    int c = 0;
    for (int i = t; i < D_; i += 256) c += (cache[i] >= th);
    redi[t] = c; __syncthreads();
    for (int st = 128; st > 0; st >>= 1) {
      if (t < st) redi[t] += redi[t + st];
      __syncthreads();
    }
    int cnt = redi[0]; __syncthreads();
    if (cnt > TOPK_K) lo = th; else hi = th;
  }
  float thr = lo;
  float amax = fmaxf(rowmax, EPS_);
  float s = 127.f / amax;
  for (int i = t; i < D_; i += 256) {
    float x = __half2float(xr[i]);
    float q = fminf(127.f, fmaxf(-128.f, roundf(x * s)));
    oq[(size_t)row * D_ + i] = (cache[i] >= thr) ? (int8_t)q : (int8_t)0;
  }
  if (t == 0) rscale_o[row] = amax / 127.f;
}

__global__ __launch_bounds__(256)
void k_gemm_out(const int8_t* oq, const float* rscale_o, const int8_t* wo_t,
                const float* wscale, float* out) {
  float ws = wscale[3];
  int wave = threadIdx.x >> 5;
  int lane = threadIdx.x & 31;
  int m0 = blockIdx.y * 16;
  int n0 = blockIdx.x * 128 + wave * 16;
  v8i acc = {};
  for (int k0 = 0; k0 < D_; k0 += 64) {
    v8i a = fragA8(oq,   D_, m0, k0);
    v8i b = fragB8(wo_t, D_, n0, k0);
    acc = wmma_iu8(a, b, acc);
  }
  int hf = lane >> 4;
  int n = n0 + (lane & 15);
  for (int r = 0; r < 8; ++r) {
    int m = m0 + r + 8 * hf;
    out[(size_t)m * D_ + n] = (float)acc[r] * rscale_o[m] * ws;
  }
}

extern "C" void kernel_launch(void* const* d_in, const int* in_sizes, int n_in,
                              void* d_out, int out_size, void* d_ws, size_t ws_size,
                              hipStream_t stream) {
  (void)in_sizes; (void)n_in; (void)out_size; (void)ws_size;
  const float* x  = (const float*)d_in[0];
  const float* wq = (const float*)d_in[1];
  const float* wk = (const float*)d_in[2];
  const float* wv = (const float*)d_in[3];
  const float* wo = (const float*)d_in[4];
  float* out = (float*)d_out;

  char* ws = (char*)d_ws;
  size_t off = 0;
  auto alloc = [&](size_t bytes) -> char* {
    char* p = ws + off;
    off = (off + bytes + 255) & ~(size_t)255;
    return p;
  };
  uint8_t* xq       = (uint8_t*)alloc((size_t)NROWS * D_);
  float*   rscale_x = (float*)  alloc((size_t)NROWS * sizeof(float));
  uint8_t* wt_fp8   = (uint8_t*)alloc((size_t)4 * D_ * D_);
  int8_t*  wt_i8    = (int8_t*) alloc((size_t)4 * D_ * D_);
  float*   wsum     = (float*)  alloc(4 * sizeof(float));
  float*   wscale   = (float*)  alloc(4 * sizeof(float));
  __half*  qh       = (__half*) alloc((size_t)NROWS * D_ * 2);
  __half*  kh       = (__half*) alloc((size_t)NROWS * D_ * 2);
  __half*  vh       = (__half*) alloc((size_t)NROWS * D_ * 2);
  __half*  ohb      = (__half*) alloc((size_t)NROWS * D_ * 2);
  int8_t*  oq       = (int8_t*) alloc((size_t)NROWS * D_);
  float*   rscale_o = (float*)  alloc((size_t)NROWS * sizeof(float));

  k_zero<<<1, 32, 0, stream>>>(wsum, 4);
  k_abssum<<<dim3(256, 1, 4), 256, 0, stream>>>(wq, wk, wv, wo, wsum);
  k_ternarize<<<dim3(1024, 1, 4), 256, 0, stream>>>(wq, wk, wv, wo, wsum,
                                                    wt_fp8, wt_i8, wscale);
  k_quant_x<<<NROWS, 256, 0, stream>>>(x, xq, rscale_x);
  k_gemm_qkv<<<dim3(D_ / 128, NROWS / 16, 3), 256, 0, stream>>>(
      xq, rscale_x, wt_fp8, wscale, qh, kh, vh);
  k_rope<<<dim3((NROWS * H_ * 64) / 256, 2), 256, 0, stream>>>(qh, kh);
  k_attn<<<dim3(B_ * H_, T_ / 64), 128, 0, stream>>>(qh, kh, vh, ohb);
  k_topk_quant<<<NROWS, 256, 0, stream>>>(ohb, oq, rscale_o);
  k_gemm_out<<<dim3(D_ / 128, NROWS / 16), 256, 0, stream>>>(
      oq, rscale_o, wt_i8 + (size_t)3 * D_ * D_, wscale, out);
}
